// GRU2_51505247813884
// MI455X (gfx1250) — matrix operand order repaired and checked by reading, not compiled, not run on hardware
//
#include <hip/hip_runtime.h>
#include <hip/hip_bf16.h>
#include <hip/hip_fp16.h>

// ---------------------------------------------------------------------------
// Bidirectional GRU, B=64 T=512 D=H=512, MI455X (gfx1250, wave32, WMMA).
//  Phase 1: fp32 -> f16 conversion of x and the four weight matrices.
//  Phase 2: zi = x @ w_ih^T + b_ih (both dirs), v_wmma_f32_16x16x32_f16.
//           Per-wave serialization hidden by 8+ waves/SIMD occupancy.
//  Phase 3: persistent recurrence: 32 blocks/direction. w_hh fragments
//           REGISTER-RESIDENT for all 512 steps; each step loads all 16
//           h_prev A-fragments up front, then the 48-WMMA chain. Per-step
//           sync is a DISTRIBUTED FLAG barrier: one release store per block
//           (no contended atomic RMW), 32 lanes acquire-poll one flag each
//           in parallel. zi gate inputs for s+1 prefetch before the barrier.
// ---------------------------------------------------------------------------

typedef __attribute__((ext_vector_type(16))) _Float16 v16h;
typedef __attribute__((ext_vector_type(8)))  _Float16 v8h;
typedef __attribute__((ext_vector_type(4)))  _Float16 v4h;
typedef __attribute__((ext_vector_type(8)))  float    v8f;

union FragU { v16h v; v8h h[2]; };

// A fragment (16x32 f16, M x K): lane = (m%16) + 16*kchunk.
static __device__ __forceinline__ v16h load_a_frag(const _Float16* p) {
  FragU f;
  f.h[0] = *(const v8h*)(p);
  f.h[1] = *(const v8h*)(p + 8);
  return f.v;
}
// B fragment (32x16 f16, K x N): lane = n + 16*kchunk; 16 contiguous K-halves
// of weight row n (weights row-major over K == column-major B operand).
static __device__ __forceinline__ v16h load_b_frag(const _Float16* p) {
  FragU f;
  f.h[0] = *(const v8h*)(p);
  f.h[1] = *(const v8h*)(p + 8);
  return f.v;
}

static __device__ __forceinline__ float sigm(float x) {
  return 1.0f / (1.0f + __expf(-x));
}
static __device__ __forceinline__ float tanh_fast(float x) {
  return 2.0f / (1.0f + __expf(-2.0f * x)) - 1.0f;
}

// ---------------------------------------------------------------------------
__global__ __launch_bounds__(256) void f32_to_f16_vec4(
    const float* __restrict__ src, _Float16* __restrict__ dst, int n4) {
  int i = blockIdx.x * blockDim.x + threadIdx.x;
  if (i < n4) {
    float4 v = ((const float4*)src)[i];
    v4h o = { (_Float16)v.x, (_Float16)v.y, (_Float16)v.z, (_Float16)v.w };
    ((v4h*)dst)[i] = o;
  }
}

// ---------------------------------------------------------------------------
// zi GEMM: grid (512, 24, 2); block 128 (4 waves). Wave: one 16-row M-tile x
// four 16-col N-tiles, K=512 in 16 WMMA steps. Load stalls are hidden across
// the 8+ co-resident waves per SIMD (compiler keeps VGPRs low here).
__global__ __launch_bounds__(128) void zi_gemm_kernel(
    const _Float16* __restrict__ xh,   // (32768, 512)
    const _Float16* __restrict__ wih,  // 2 x (1536, 512)
    const float* __restrict__ bih_fw,
    const float* __restrict__ bih_bw,
    _Float16* __restrict__ zi)         // 2 x (512, 64, 1536)
{
  const int wave  = threadIdx.x >> 5;
  const int lane  = threadIdx.x & 31;
  const int laneM = lane & 15;
  const int laneC = lane >> 4;
  const int dir   = blockIdx.z;
  const int m0    = blockIdx.x * 64 + wave * 16;
  const int n0    = blockIdx.y * 64;

  const _Float16* w   = wih + (size_t)dir * (1536 * 512);
  const float*    bih = dir ? bih_bw : bih_fw;
  _Float16*       zo  = zi + (size_t)dir * ((size_t)512 * 64 * 1536);

  const _Float16* aBase = xh + (size_t)(m0 + laneM) * 512 + laneC * 16;
  const _Float16* bBase = w  + (size_t)(n0 + laneM) * 512 + laneC * 16;

  v8f acc[4] = {};

  v16h a = load_a_frag(aBase);
  v16h b[4];
#pragma unroll
  for (int j = 0; j < 4; ++j) b[j] = load_b_frag(bBase + (size_t)j * 16 * 512);

#pragma unroll
  for (int kt = 0; kt < 16; ++kt) {
    v16h an;
    v16h bn[4];
    if (kt < 15) {
      an = load_a_frag(aBase + (kt + 1) * 32);
#pragma unroll
      for (int j = 0; j < 4; ++j)
        bn[j] = load_b_frag(bBase + (size_t)j * 16 * 512 + (kt + 1) * 32);
    }
#pragma unroll
    for (int j = 0; j < 4; ++j)
      acc[j] = __builtin_amdgcn_wmma_f32_16x16x32_f16(
          false, a, false, b[j], (short)0, acc[j], false, false);
    if (kt < 15) {
      a = an;
#pragma unroll
      for (int j = 0; j < 4; ++j) b[j] = bn[j];
    }
  }

  const int mrow0 = m0 + laneC * 8;  // D rows: mrow0 + v
#pragma unroll
  for (int j = 0; j < 4; ++j) {
    const int   ncol = n0 + j * 16 + laneM;
    const float bias = bih[ncol];
#pragma unroll
    for (int v = 0; v < 8; ++v) {
      int m = mrow0 + v;
      int b2 = m >> 9;     // batch
      int t  = m & 511;    // time
      zo[(size_t)(t * 64 + b2) * 1536 + ncol] = (_Float16)(acc[j][v] + bias);
    }
  }
}

// ---------------------------------------------------------------------------
// Persistent recurrence. 64 blocks: [0,32) forward, [32,64) backward.
// Block owns H-columns [c0, c0+16) of ALL THREE gates -> gate math is local.
// Wave w handles batch rows [16w, 16w+16). The 48 w_hh B-fragments (3 gates x
// 16 k-steps, 384 VGPRs) are loaded ONCE and stay register-resident for all
// 512 steps. ~600 VGPRs/wave; launch_bounds(128,1) allows 1 wave/EU so the
// allocator does not spill to chase occupancy.
__global__ __launch_bounds__(128, 1) void gru_recurrence_kernel(
    const _Float16* __restrict__ zi,   // 2 x (512, 64, 1536)
    const _Float16* __restrict__ whh,  // 2 x (1536, 512)
    const float* __restrict__ bhh_fw,
    const float* __restrict__ bhh_bw,
    _Float16* __restrict__ hbuf,       // 2 dirs x 2 bufs x (64, 512)
    unsigned* __restrict__ flags,      // 2 x 32 monotonic step flags (zeroed)
    float* __restrict__ out)           // (64,512,1024) then h_fw, h_bw
{
  const int lane  = threadIdx.x & 31;
  const int wave  = threadIdx.x >> 5;
  const int laneM = lane & 15;
  const int laneC = lane >> 4;
  const int dir   = blockIdx.x >> 5;   // 32 blocks per direction
  const int me    = blockIdx.x & 31;   // block id within direction
  const int c0    = me * 16;

  const float*    bhh = dir ? bhh_bw : bhh_fw;
  const _Float16* zid = zi  + (size_t)dir * ((size_t)512 * 64 * 1536);
  const _Float16* w   = whh + (size_t)dir * (1536 * 512);

  const int m0  = wave * 16;
  const int mA  = m0 + laneM;          // A-fragment row
  const int mD0 = m0 + laneC * 8;      // first D row for this lane
  const int j   = c0 + laneM;          // column within H

  const float bR = bhh[j], bZ = bhh[512 + j], bN = bhh[1024 + j];

  // -------- load all w_hh B-fragments into registers (loop-invariant) -----
  const _Float16* wR = w + (size_t)(       c0 + laneM) * 512 + laneC * 16;
  const _Float16* wZ = w + (size_t)( 512 + c0 + laneM) * 512 + laneC * 16;
  const _Float16* wN = w + (size_t)(1024 + c0 + laneM) * 512 + laneC * 16;
  v16h fR[16], fZ[16], fN[16];
#pragma unroll
  for (int kt = 0; kt < 16; ++kt) {
    fR[kt] = load_b_frag(wR + kt * 32);
    fZ[kt] = load_b_frag(wZ + kt * 32);
    fN[kt] = load_b_frag(wN + kt * 32);
  }

  unsigned* myFlags = flags + dir * 32;  // this direction's 32 flag words

  // zi gate inputs for the first step (later steps prefetch before the
  // barrier so the loads complete during the spin).
  _Float16 zR[8], zZ[8], zN[8];
  {
    const int t0 = dir ? 511 : 0;
    const _Float16* zt = zid + (size_t)t0 * (64 * 1536);
#pragma unroll
    for (int v = 0; v < 8; ++v) {
      int m = mD0 + v;
      zR[v] = zt[(size_t)m * 1536 + j];
      zZ[v] = zt[(size_t)m * 1536 + 512 + j];
      zN[v] = zt[(size_t)m * 1536 + 1024 + j];
    }
  }

  for (int s = 0; s < 512; ++s) {
    const int t = dir ? (511 - s) : s;
    const _Float16* hprev = hbuf + (size_t)(dir * 2 + (s & 1)) * (64 * 512);
    _Float16*       hnext = hbuf + (size_t)(dir * 2 + ((s + 1) & 1)) * (64 * 512);

    // issue the h_prev gate-math loads early; they resolve under the WMMAs
    float hp[8];
#pragma unroll
    for (int v = 0; v < 8; ++v)
      hp[v] = (float)hprev[(size_t)(mD0 + v) * 512 + j];

    // issue ALL A-fragment loads for this step up front (in-order retirement
    // keeps the WMMA chain on partial s_wait_loadcnt values)
    const _Float16* aB = hprev + (size_t)mA * 512 + laneC * 16;
    v16h aF[16];
#pragma unroll
    for (int kt = 0; kt < 16; ++kt) aF[kt] = load_a_frag(aB + kt * 32);

    v8f accR = {}, accZ = {}, accN = {};
#pragma unroll
    for (int kt = 0; kt < 16; ++kt) {
      accR = __builtin_amdgcn_wmma_f32_16x16x32_f16(false, aF[kt], false, fR[kt], (short)0, accR, false, false);
      accZ = __builtin_amdgcn_wmma_f32_16x16x32_f16(false, aF[kt], false, fZ[kt], (short)0, accZ, false, false);
      accN = __builtin_amdgcn_wmma_f32_16x16x32_f16(false, aF[kt], false, fN[kt], (short)0, accN, false, false);
    }

    // gate math + h update for columns [c0, c0+16)
#pragma unroll
    for (int v = 0; v < 8; ++v) {
      int   m  = mD0 + v;
      float r  = sigm((float)zR[v] + accR[v] + bR);
      float z  = sigm((float)zZ[v] + accZ[v] + bZ);
      float n  = tanh_fast((float)zN[v] + r * (accN[v] + bN));
      float hn = (1.0f - z) * n + z * hp[v];
      hnext[(size_t)m * 512 + j] = (_Float16)hn;
      out[((size_t)m * 512 + t) * 1024 + dir * 512 + j] = hn;
      if (s == 511)  // final hidden state outputs
        out[(size_t)64 * 512 * 1024 + (size_t)dir * (64 * 512) + (size_t)m * 512 + j] = hn;
    }

    // prefetch next step's zi gate inputs (independent of h) so the loads
    // fly during the barrier wait
    if (s < 511) {
      const int tn = dir ? (511 - (s + 1)) : (s + 1);
      const _Float16* zt = zid + (size_t)tn * (64 * 1536);
#pragma unroll
      for (int v = 0; v < 8; ++v) {
        int m = mD0 + v;
        zR[v] = zt[(size_t)m * 1536 + j];
        zZ[v] = zt[(size_t)m * 1536 + 512 + j];
        zN[v] = zt[(size_t)m * 1536 + 1024 + j];
      }
    }

    // --- distributed flag barrier (no contended atomic RMW) ---------------
    // Each thread makes its h-writes agent-visible, then one release store
    // publishes this block's arrival; 32 lanes acquire-poll one flag each in
    // parallel. Monotonic flags (>= s+1) need no reset; a block can be at
    // most one step ahead, so wraparound hazards cannot occur.
    __threadfence();
    __syncthreads();
    if (threadIdx.x == 0) {
      __hip_atomic_store(&myFlags[me], (unsigned)(s + 1),
                         __ATOMIC_RELEASE, __HIP_MEMORY_SCOPE_AGENT);
    }
    if (threadIdx.x < 32) {
      while (__hip_atomic_load(&myFlags[threadIdx.x],
                               __ATOMIC_ACQUIRE, __HIP_MEMORY_SCOPE_AGENT)
             < (unsigned)(s + 1)) {
        __builtin_amdgcn_s_sleep(1);
      }
    }
    __syncthreads();
    __threadfence();
  }
}

// ---------------------------------------------------------------------------
extern "C" void kernel_launch(void* const* d_in, const int* in_sizes, int n_in,
                              void* d_out, int out_size, void* d_ws, size_t ws_size,
                              hipStream_t stream) {
  (void)in_sizes; (void)n_in; (void)out_size; (void)ws_size;

  const float* x      = (const float*)d_in[0];
  const float* wih_fw = (const float*)d_in[1];
  const float* whh_fw = (const float*)d_in[2];
  const float* bih_fw = (const float*)d_in[3];
  const float* bhh_fw = (const float*)d_in[4];
  const float* wih_bw = (const float*)d_in[5];
  const float* whh_bw = (const float*)d_in[6];
  const float* bih_bw = (const float*)d_in[7];
  const float* bhh_bw = (const float*)d_in[8];

  // Workspace layout (~230 MB total)
  char*  ws  = (char*)d_ws;
  size_t off = 0;
  _Float16* xh    = (_Float16*)(ws + off); off += (size_t)64 * 512 * 512 * 2;       // 32 MB
  _Float16* wihh  = (_Float16*)(ws + off); off += (size_t)2 * 1536 * 512 * 2;       // 3 MB
  _Float16* whhh  = (_Float16*)(ws + off); off += (size_t)2 * 1536 * 512 * 2;       // 3 MB
  _Float16* zi    = (_Float16*)(ws + off); off += (size_t)2 * 512 * 64 * 1536 * 2;  // 192 MB
  _Float16* hbuf  = (_Float16*)(ws + off); off += (size_t)2 * 2 * 64 * 512 * 2;     // 256 KB
  unsigned* flags = (unsigned*)(ws + off); off += (size_t)2 * 32 * 4;               // 256 B

  // Zero h0 (both dirs, both buffers) and the barrier flags (contiguous).
  hipMemsetAsync(hbuf, 0, (size_t)2 * 2 * 64 * 512 * 2 + (size_t)2 * 32 * 4, stream);

  // fp32 -> f16 conversions
  {
    int n4 = (64 * 512 * 512) / 4;
    f32_to_f16_vec4<<<(n4 + 255) / 256, 256, 0, stream>>>(x, xh, n4);
    n4 = (1536 * 512) / 4;
    f32_to_f16_vec4<<<(n4 + 255) / 256, 256, 0, stream>>>(wih_fw, wihh, n4);
    f32_to_f16_vec4<<<(n4 + 255) / 256, 256, 0, stream>>>(wih_bw, wihh + 1536 * 512, n4);
    f32_to_f16_vec4<<<(n4 + 255) / 256, 256, 0, stream>>>(whh_fw, whhh, n4);
    f32_to_f16_vec4<<<(n4 + 255) / 256, 256, 0, stream>>>(whh_bw, whhh + 1536 * 512, n4);
  }

  // Input projections for both directions
  dim3 g2(512, 24, 2);
  zi_gemm_kernel<<<g2, 128, 0, stream>>>(xh, wihh, bih_fw, bih_bw, zi);

  // Persistent bidirectional recurrence (32 blocks/dir, co-resident)
  gru_recurrence_kernel<<<64, 128, 0, stream>>>(zi, whhh, bhh_fw, bhh_bw,
                                                hbuf, flags, (float*)d_out);
}
